// PROVGEM_79714593014416
// MI455X (gfx1250) — matrix operand order repaired and checked by reading, not compiled, not run on hardware
//
#include <hip/hip_runtime.h>
#include <hip/hip_bf16.h>

typedef __attribute__((ext_vector_type(16))) _Float16 v16h;
typedef __attribute__((ext_vector_type(8)))  _Float16 v8h;
typedef __attribute__((ext_vector_type(8)))  float    v8f;

constexpr int   RR = 3;
constexpr int   NN = 50000;
constexpr long long EE = 800000;
constexpr int   FF = 128;
constexpr int   DD = 128;
constexpr int   AA = 64;

// ---------------------------------------------------------------------------
// Scatter: one wave per edge. 32 lanes x 4 floats = 128 features.
// Atomic float adds land in L2 (feat + ssum fit in 192MB L2).
// ---------------------------------------------------------------------------
__launch_bounds__(256)
__global__ void scatter_kernel(const float* __restrict__ feat,
                               const int* __restrict__ src,
                               const int* __restrict__ dst,
                               float* __restrict__ ssum,
                               float* __restrict__ deg) {
    long long gid = (long long)blockIdx.x * 8 + (threadIdx.x >> 5);
    if (gid >= (long long)RR * EE) return;
    int r = (int)(gid / EE);
    long long e = gid - (long long)r * EE;
    int s = src[(long long)r * EE + e];
    int d = dst[(long long)r * EE + e];
    int lane = threadIdx.x & 31;

    const float4 v = *(const float4*)(feat + (size_t)s * FF + lane * 4);
    float* o = ssum + ((size_t)r * NN + d) * FF + lane * 4;
    __hip_atomic_fetch_add(o + 0, v.x, __ATOMIC_RELAXED, __HIP_MEMORY_SCOPE_AGENT);
    __hip_atomic_fetch_add(o + 1, v.y, __ATOMIC_RELAXED, __HIP_MEMORY_SCOPE_AGENT);
    __hip_atomic_fetch_add(o + 2, v.z, __ATOMIC_RELAXED, __HIP_MEMORY_SCOPE_AGENT);
    __hip_atomic_fetch_add(o + 3, v.w, __ATOMIC_RELAXED, __HIP_MEMORY_SCOPE_AGENT);
    if (lane == 0)
        __hip_atomic_fetch_add(&deg[(size_t)r * NN + d], 1.0f,
                               __ATOMIC_RELAXED, __HIP_MEMORY_SCOPE_AGENT);
}

// ---------------------------------------------------------------------------
// h = feat + ssum / max(deg,1), cast to f16 for WMMA
// ---------------------------------------------------------------------------
__launch_bounds__(256)
__global__ void resid_kernel(const float* __restrict__ feat,
                             const float* __restrict__ ssum,
                             const float* __restrict__ deg,
                             _Float16* __restrict__ h16) {
    size_t i = (size_t)blockIdx.x * 256 + threadIdx.x;  // over R*N*F
    if (i >= (size_t)RR * NN * FF) return;
    size_t rn = i >> 7;
    size_t f  = i & 127;
    size_t n  = rn % NN;
    float dg = fmaxf(deg[rn], 1.0f);
    h16[i] = (_Float16)(feat[n * FF + f] + ssum[i] / dg);
}

// ---------------------------------------------------------------------------
// Fused GEMM [N,128]@[128,128] + bias + LayerNorm + ReLU via v_wmma_f32_16x16x32_f16
// Block: 256 thr = 8 waves, each wave owns 16 rows x 128 cols (8 accum tiles).
// W tile transposed into LDS as f16 so B fragments are contiguous v16h loads.
// LayerNorm done in-register via 16-lane shuffle reductions (each half-wave
// holds a full output row per accumulator slot).
// MODE 0: out16 = f16(relu(LN_r(x)))   MODE 1: outf = LN_glob(relu(LN_r(x)))
// ---------------------------------------------------------------------------
template <int MODE>
__launch_bounds__(256)
__global__ void gemm_ln_kernel(const _Float16* __restrict__ Aall,  // [R,N,128] f16
                               const float* __restrict__ Wall,     // [R,128,128] f32
                               const float* __restrict__ bias,     // [R,128]
                               const float* __restrict__ lng,      // [R,128]
                               const float* __restrict__ lnb,      // [R,128]
                               const float* __restrict__ LNg,      // [128]  (MODE 1)
                               const float* __restrict__ LNb,      // [128]  (MODE 1)
                               _Float16* __restrict__ out16,       // MODE 0
                               float* __restrict__ outf)           // MODE 1
{
    __shared__ _Float16 sWt[128 * 128];  // transposed: sWt[col*128 + k], 32KB

    const int r    = blockIdx.y;
    const int row0 = blockIdx.x * 128;
    const int tid  = threadIdx.x;

    // load W[r] (f32, row-major [k][col]) -> LDS f16 transposed [col][k]
    const float* Wg = Wall + (size_t)r * 128 * 128;
    for (int idx = tid; idx < 128 * 128; idx += 256) {
        int k = idx >> 7, col = idx & 127;
        sWt[col * 128 + k] = (_Float16)Wg[idx];
    }
    __syncthreads();

    const int wave = tid >> 5;
    const int lane = tid & 31;
    const int l16  = lane & 15;
    const int hi   = lane >> 4;  // half-wave select

    // A row this lane streams (ISA 16-bit A 16x32 layout: lanes 0-15 M=0..15)
    const int mrow = row0 + wave * 16 + l16;
    const bool ainb = (mrow < NN);
    const _Float16* Arow = Aall + ((size_t)r * NN + (ainb ? mrow : 0)) * 128;

    v8f c[8];
#pragma unroll
    for (int t = 0; t < 8; t++)
#pragma unroll
        for (int i = 0; i < 8; i++) c[t][i] = 0.0f;

#pragma unroll
    for (int k0 = 0; k0 < 128; k0 += 32) {
        // A fragment: halves 0-15 -> K=k0+hi*16..+15 (matches B)
        v16h a;
        v8h alo = *(const v8h*)(Arow + k0 + hi * 16);
        v8h ahi = *(const v8h*)(Arow + k0 + hi * 16 + 8);
#pragma unroll
        for (int i = 0; i < 8; i++) { a[i] = alo[i]; a[8 + i] = ahi[i]; }
        if (!ainb) {
#pragma unroll
            for (int i = 0; i < 16; i++) a[i] = (_Float16)0.0f;
        }
#pragma unroll
        for (int nt = 0; nt < 8; nt++) {
            // B fragment (32x16): lane half 0 holds K=k0..k0+15, half 1 K=k0+16..+31
            const int col = nt * 16 + l16;
            v16h b = *(const v16h*)(&sWt[col * 128 + k0 + hi * 16]);
            c[nt] = __builtin_amdgcn_wmma_f32_16x16x32_f16(
                false, a, false, b, (short)0, c[nt], false, false);
        }
    }

    // epilogue params for this lane's 8 columns (col = nt*16 + l16)
    float bias_l[8], g_l[8], bb_l[8], Gg_l[8], Gb_l[8];
    const float* Bp = bias + r * 128;
    const float* Gp = lng + r * 128;
    const float* Pp = lnb + r * 128;
#pragma unroll
    for (int nt = 0; nt < 8; nt++) {
        int col = nt * 16 + l16;
        bias_l[nt] = Bp[col];
        g_l[nt]    = Gp[col];
        bb_l[nt]   = Pp[col];
        if (MODE == 1) { Gg_l[nt] = LNg[col]; Gb_l[nt] = LNb[col]; }
    }

    // C/D layout: lane-half `hi` holds rows hi*8+v (v = accum slot), cols across
    // the 16 lanes of that half -> row reductions are 16-lane shuffles.
#pragma unroll
    for (int v = 0; v < 8; v++) {
        const int row = row0 + wave * 16 + hi * 8 + v;

        float s = 0.0f;
#pragma unroll
        for (int nt = 0; nt < 8; nt++) {
            float t = c[nt][v] + bias_l[nt];
            c[nt][v] = t;
            s += t;
        }
#pragma unroll
        for (int off = 8; off >= 1; off >>= 1) s += __shfl_xor(s, off, 32);
        const float mu = s * (1.0f / 128.0f);

        float vv = 0.0f;
#pragma unroll
        for (int nt = 0; nt < 8; nt++) {
            float t = c[nt][v] - mu;
            vv += t * t;
        }
#pragma unroll
        for (int off = 8; off >= 1; off >>= 1) vv += __shfl_xor(vv, off, 32);
        const float rs = rsqrtf(vv * (1.0f / 128.0f) + 1e-5f);

        if (MODE == 0) {
            if (row < NN) {
                _Float16* o = out16 + ((size_t)r * NN + row) * 128;
#pragma unroll
                for (int nt = 0; nt < 8; nt++) {
                    float t = fmaxf((c[nt][v] - mu) * rs * g_l[nt] + bb_l[nt], 0.0f);
                    o[nt * 16 + l16] = (_Float16)t;
                }
            }
        } else {
            // relu(LN_r) then global LN
            float s2 = 0.0f;
#pragma unroll
            for (int nt = 0; nt < 8; nt++) {
                float t = fmaxf((c[nt][v] - mu) * rs * g_l[nt] + bb_l[nt], 0.0f);
                c[nt][v] = t;
                s2 += t;
            }
#pragma unroll
            for (int off = 8; off >= 1; off >>= 1) s2 += __shfl_xor(s2, off, 32);
            const float mu2 = s2 * (1.0f / 128.0f);
            float vv2 = 0.0f;
#pragma unroll
            for (int nt = 0; nt < 8; nt++) {
                float t = c[nt][v] - mu2;
                vv2 += t * t;
            }
#pragma unroll
            for (int off = 8; off >= 1; off >>= 1) vv2 += __shfl_xor(vv2, off, 32);
            const float rs2 = rsqrtf(vv2 * (1.0f / 128.0f) + 1e-5f);
            if (row < NN) {
                float* o = outf + ((size_t)r * NN + row) * 128;
#pragma unroll
                for (int nt = 0; nt < 8; nt++)
                    o[nt * 16 + l16] = (c[nt][v] - mu2) * rs2 * Gg_l[nt] + Gb_l[nt];
            }
        }
    }
}

// ---------------------------------------------------------------------------
// e[r,n] = Wa2[r] . tanh(hfin[r,n,:] @ Wa1[r])   (one wave per (r,n))
// ---------------------------------------------------------------------------
__launch_bounds__(256)
__global__ void attn_score_kernel(const float* __restrict__ hfin,  // [R,N,128]
                                  const float* __restrict__ Wa1,   // [R,128,64]
                                  const float* __restrict__ Wa2,   // [R,64]
                                  float* __restrict__ e)           // [R,N]
{
    __shared__ float sh[8][128];
    long long gid = (long long)blockIdx.x * 8 + (threadIdx.x >> 5);
    const int w = threadIdx.x >> 5;
    const int lane = threadIdx.x & 31;
    const bool act = gid < (long long)RR * NN;
    int r = 0;
    long long n = 0;
    if (act) { r = (int)(gid / NN); n = gid - (long long)r * NN; }

    if (act) {
        const float4 v = *(const float4*)(hfin + ((size_t)r * NN + n) * 128 + lane * 4);
        sh[w][lane * 4 + 0] = v.x;
        sh[w][lane * 4 + 1] = v.y;
        sh[w][lane * 4 + 2] = v.z;
        sh[w][lane * 4 + 3] = v.w;
    }
    __syncthreads();

    if (act) {
        const float* Wp = Wa1 + (size_t)r * 128 * 64;
        float acc0 = 0.0f, acc1 = 0.0f;
        for (int d = 0; d < 128; d++) {
            const float hv = sh[w][d];
            acc0 += hv * Wp[d * 64 + lane];
            acc1 += hv * Wp[d * 64 + 32 + lane];
        }
        float p = tanhf(acc0) * Wa2[r * 64 + lane] + tanhf(acc1) * Wa2[r * 64 + 32 + lane];
#pragma unroll
        for (int off = 16; off >= 1; off >>= 1) p += __shfl_xor(p, off, 32);
        if (lane == 0) e[(size_t)r * NN + n] = p;
    }
}

// ---------------------------------------------------------------------------
// out[n,d] = sum_r softmax_r(e[:,n]) * hfin[r,n,d]
// ---------------------------------------------------------------------------
__launch_bounds__(256)
__global__ void combine_kernel(const float* __restrict__ hfin,
                               const float* __restrict__ e,
                               float* __restrict__ out) {
    size_t i = (size_t)blockIdx.x * 256 + threadIdx.x;  // over N*128
    if (i >= (size_t)NN * DD) return;
    size_t n = i >> 7;
    float e0 = e[n], e1 = e[(size_t)NN + n], e2 = e[(size_t)2 * NN + n];
    float m = fmaxf(e0, fmaxf(e1, e2));
    float w0 = __expf(e0 - m), w1 = __expf(e1 - m), w2 = __expf(e2 - m);
    float inv = 1.0f / (w0 + w1 + w2);
    out[i] = (w0 * hfin[i] + w1 * hfin[(size_t)NN * DD + i] +
              w2 * hfin[(size_t)2 * NN * DD + i]) * inv;
}

// ---------------------------------------------------------------------------
extern "C" void kernel_launch(void* const* d_in, const int* in_sizes, int n_in,
                              void* d_out, int out_size, void* d_ws, size_t ws_size,
                              hipStream_t stream) {
    const float* feat = (const float*)d_in[0];
    const int*   src  = (const int*)d_in[1];
    const int*   dst  = (const int*)d_in[2];
    const float* W0   = (const float*)d_in[3];
    const float* b0   = (const float*)d_in[4];
    const float* W1   = (const float*)d_in[5];
    const float* b1   = (const float*)d_in[6];
    const float* ln_g = (const float*)d_in[7];
    const float* ln_b = (const float*)d_in[8];
    const float* LN_g = (const float*)d_in[9];
    const float* LN_b = (const float*)d_in[10];
    const float* Wa1  = (const float*)d_in[11];
    const float* Wa2  = (const float*)d_in[12];
    float* out = (float*)d_out;

    char* ws = (char*)d_ws;
    size_t off = 0;
    auto alloc = [&](size_t bytes) -> void* {
        void* p = ws + off;
        off += (bytes + 255) & ~(size_t)255;
        return p;
    };
    float*     ssum = (float*)alloc((size_t)RR * NN * FF * sizeof(float));  // later: hfin
    float*     deg  = (float*)alloc((size_t)RR * NN * sizeof(float));       // later: e
    _Float16*  h16  = (_Float16*)alloc((size_t)RR * NN * FF * sizeof(_Float16));
    _Float16*  x16  = (_Float16*)alloc((size_t)RR * NN * DD * sizeof(_Float16));
    float* hfin = ssum;  // ssum dead after resid_kernel
    float* evec = deg;   // deg dead after resid_kernel

    hipMemsetAsync(ssum, 0, (size_t)RR * NN * FF * sizeof(float), stream);
    hipMemsetAsync(deg, 0, (size_t)RR * NN * sizeof(float), stream);

    // per-relation mean aggregation (atomic scatter-add + degree)
    long long totE = (long long)RR * EE;
    scatter_kernel<<<(unsigned)((totE + 7) / 8), 256, 0, stream>>>(feat, src, dst, ssum, deg);

    // residual add + f16 cast
    size_t totH = (size_t)RR * NN * FF;
    resid_kernel<<<(unsigned)((totH + 255) / 256), 256, 0, stream>>>(feat, ssum, deg, h16);

    // two fused WMMA GEMM + LN + ReLU stages
    dim3 gg((NN + 127) / 128, RR);
    gemm_ln_kernel<0><<<gg, 256, 0, stream>>>(h16, W0, b0, ln_g, ln_b,
                                              nullptr, nullptr, x16, nullptr);
    gemm_ln_kernel<1><<<gg, 256, 0, stream>>>(x16, W1, b1, ln_g, ln_b,
                                              LN_g, LN_b, nullptr, hfin);

    // semantic attention + combine
    long long totRN = (long long)RR * NN;
    attn_score_kernel<<<(unsigned)((totRN + 7) / 8), 256, 0, stream>>>(hfin, Wa1, Wa2, evec);
    size_t totO = (size_t)NN * DD;
    combine_kernel<<<(unsigned)((totO + 255) / 256), 256, 0, stream>>>(hfin, evec, out);
}
